// GMMAttention_36962488549696
// MI455X (gfx1250) — matrix operand order repaired and checked by reading, not compiled, not run on hardware
//
#include <hip/hip_runtime.h>
#include <hip/hip_bf16.h>
#include <stdint.h>

// ---------------------------------------------------------------------------
// Problem dims
// ---------------------------------------------------------------------------
#define Bsz   128
#define Nseq  197
#define Cdim  768
#define Hhead 12
#define Dhead 64
#define Mrows (Bsz * Nseq)          // 25216, divisible by 128
#define NPAD  208                   // 13 tiles of 16 (padded sequence)
#define VPAD  224                   // 7 K-fragments of 32 (padded P / V^T row)
#define KHALF 384                   // K staged into LDS in two halves

typedef __attribute__((ext_vector_type(16))) __bf16   v16bf;
typedef __attribute__((ext_vector_type(8)))  __bf16   v8bf;
typedef __attribute__((ext_vector_type(8)))  float    v8f;
typedef __attribute__((ext_vector_type(4)))  uint32_t v4u;
typedef __attribute__((ext_vector_type(8)))  int      v8i;
typedef __attribute__((ext_vector_type(4)))  int      v4i;

// ---------------------------------------------------------------------------
// WMMA helpers (wave32; A/B use this file's contiguous k order, C/D per CDNA5 ISA §7.12.2)
// A fragment 16x32 bf16: lane l holds row m=l&15; halfwords cover
//   K = (l>>4)*16 .. +15 -> one contiguous 32B chunk (matches B).
// B fragment 32x16 bf16 from an N-major (N x K row-major) operand:
//   lane l holds col n=l&15, K = (l>>4)*16 .. +15 -> one contiguous 32B chunk.
// C/D 16x16 f32: VGPR r holds (row r + 8*(l>>4), col l&15).
// ---------------------------------------------------------------------------
__device__ __forceinline__ v16bf load_a_frag(const __bf16* base, int lda,
                                             int row0, int kbase) {
    const int l    = threadIdx.x & 31;
    const int m    = l & 15;
    const int koff = (l >> 4) * 16;
    const __bf16* p = base + (size_t)(row0 + m) * lda + kbase + koff;
    v8bf lo = *(const v8bf*)(p);
    v8bf hi = *(const v8bf*)(p + 8);
    return __builtin_shufflevector(lo, hi, 0, 1, 2, 3, 4, 5, 6, 7,
                                           8, 9, 10, 11, 12, 13, 14, 15);
}

__device__ __forceinline__ v16bf load_b_frag(const __bf16* ntk, int ldb,
                                             int col0, int kbase) {
    const int l    = threadIdx.x & 31;
    const int n    = l & 15;
    const int koff = (l >> 4) * 16;
    return *(const v16bf*)(ntk + (size_t)(col0 + n) * ldb + kbase + koff);
}

__device__ __forceinline__ v8f wmma_bf16(v16bf a, v16bf b, v8f c) {
    return __builtin_amdgcn_wmma_f32_16x16x32_bf16(false, a, false, b,
                                                   (short)0, c, false, false);
}

// ---------------------------------------------------------------------------
// TDM: 2D tensor tile (bf16) -> LDS.  Descriptor per CDNA5 ISA:
//   g0: [1:0]=count=1, [63:32]=lds_addr, [120:64]=global_addr, [127:126]=2
//   g1: [17:16]=data_size(1 -> 2B), [79:48]=tensor_dim0, [111:80]=tensor_dim1,
//       [127:112]=tile_dim0, [143:128]=tile_dim1, [207:160]=dim0_stride
// Issued by one thread (TDM ignores EXEC); tracked with TENSORcnt.
// 6-arg builtin form (clang-23 / therock-10.0 headers).
// ---------------------------------------------------------------------------
__device__ __forceinline__ void tdm_load_2d(const __bf16* gsrc, uint32_t ldsOff,
                                            int tileK, int tileRows,
                                            int tensorK, int tensorRows,
                                            int rowStride) {
    uint64_t ga = (uint64_t)(uintptr_t)gsrc;
    v4u g0;
    g0[0] = 1u;                                               // count=1
    g0[1] = ldsOff;                                           // lds_addr (bytes)
    g0[2] = (uint32_t)ga;                                     // global_addr lo
    g0[3] = ((uint32_t)(ga >> 32) & 0x01FFFFFFu) | (2u << 30);// addr hi | type=2
    v8i g1;
    g1[0] = (int)(1u << 16);                                  // data_size = 2B
    g1[1] = (int)(((uint32_t)tensorK & 0xFFFFu) << 16);       // dim0 lo16
    g1[2] = (int)(((uint32_t)tensorK >> 16) |
                  (((uint32_t)tensorRows & 0xFFFFu) << 16));  // dim0 hi | dim1 lo
    g1[3] = (int)(((uint32_t)tensorRows >> 16) |
                  (((uint32_t)tileK & 0xFFFFu) << 16));       // dim1 hi | tile0
    g1[4] = (int)((uint32_t)tileRows & 0xFFFFu);              // tile1; tile2=0
    g1[5] = (int)(uint32_t)rowStride;                         // dim0_stride lo32
    g1[6] = 0;                                                // stride hi, d1s lo
    g1[7] = 0;
    v4i z4 = {0, 0, 0, 0};
    v8i z8 = {0, 0, 0, 0, 0, 0, 0, 0};
    __builtin_amdgcn_tensor_load_to_lds(g0, g1, z4, z4, z8, 0);
}

// ---------------------------------------------------------------------------
// Elementwise helpers
// ---------------------------------------------------------------------------
__global__ void cvt_f32_to_bf16(const float* __restrict__ src,
                                __bf16* __restrict__ dst, int n) {
    int i = blockIdx.x * blockDim.x + threadIdx.x;
    if (i < n) dst[i] = (__bf16)src[i];
}

__global__ void zero_b128(uint4* __restrict__ p, long n) {
    long i = (long)blockIdx.x * blockDim.x + threadIdx.x;
    if (i < n) p[i] = make_uint4(0u, 0u, 0u, 0u);
}

// ---------------------------------------------------------------------------
// Kernel 1: QKV projection  out[m, s*768 + h*64 + d] = sum_k x[m,k]*W[c,k]
// 256 threads = 8 waves; WG tile 128x64; wave tile 32x32 (2x2 WMMA tiles).
// Weight tile (64 x 768) staged in LDS by the TDM in two 384-wide K halves.
// Scatters into padded Q[B,H,208,64], K[B,H,208,64], V^T[B,H,64,224] (bf16).
// ---------------------------------------------------------------------------
__global__ __launch_bounds__(256) void gemm_qkv(
    const __bf16* __restrict__ xb,   // [25216][768]
    const __bf16* __restrict__ wb,   // [2304][768]  (N x K major already)
    __bf16* __restrict__ qd, __bf16* __restrict__ kd, __bf16* __restrict__ vt) {
    __shared__ __align__(16) __bf16 wtile[64 * KHALF];        // 48 KB

    const int lane    = threadIdx.x & 31;
    const int wave    = threadIdx.x >> 5;
    const int rowBase = blockIdx.x * 128 + (wave & 3) * 32;
    const int colBase = blockIdx.y * 64 + (wave >> 2) * 32;
    const int colLoc  = (wave >> 2) * 32;

    v8f c00 = {}, c01 = {}, c10 = {}, c11 = {};
    for (int half = 0; half < 2; ++half) {
        if (threadIdx.x == 0)
            tdm_load_2d(wb + (size_t)(blockIdx.y * 64) * Cdim + half * KHALF,
                        (uint32_t)(uintptr_t)(void*)wtile,
                        KHALF, 64, Cdim, 64, Cdim);
        if (threadIdx.x < 32) __builtin_amdgcn_s_wait_tensorcnt(0);
        __syncthreads();
        for (int kb = 0; kb < KHALF; kb += 32) {
            v16bf a0 = load_a_frag(xb, Cdim, rowBase, half * KHALF + kb);
            v16bf a1 = load_a_frag(xb, Cdim, rowBase + 16, half * KHALF + kb);
            v16bf b0 = load_b_frag(wtile, KHALF, colLoc, kb);
            v16bf b1 = load_b_frag(wtile, KHALF, colLoc + 16, kb);
            c00 = wmma_bf16(a0, b0, c00);
            c01 = wmma_bf16(a0, b1, c01);
            c10 = wmma_bf16(a1, b0, c10);
            c11 = wmma_bf16(a1, b1, c11);
        }
        __syncthreads();
    }

    const int rOff = 8 * (lane >> 4);
    const int nCol = lane & 15;
    v8f acc[4] = {c00, c01, c10, c11};
#pragma unroll
    for (int t = 0; t < 4; ++t) {
        const int rt = t >> 1, ct = t & 1;
#pragma unroll
        for (int r = 0; r < 8; ++r) {
            int R  = rowBase + rt * 16 + r + rOff;
            int Cc = colBase + ct * 16 + nCol;
            int b  = R / Nseq, n = R % Nseq;
            int s  = Cc / Cdim, rem = Cc % Cdim;
            int h  = rem >> 6, d = rem & 63;
            __bf16 val = (__bf16)acc[t][r];
            size_t bh = (size_t)(b * Hhead + h);
            if (s == 0)       qd[(bh * NPAD + n) * Dhead + d] = val;
            else if (s == 1)  kd[(bh * NPAD + n) * Dhead + d] = val;
            else              vt[(bh * Dhead + d) * VPAD + n] = val;
        }
    }
}

// ---------------------------------------------------------------------------
// Kernel 2: fused GMM attention.  grid = (B*H, 4), block = 128 (4 waves).
// Each wave owns one 16-row query tile; K loop over 13 key tiles; P tile
// staged bf16 in LDS (stride VPAD) and fed back as WMMA A-fragments for P*V.
// O is normalized by the accumulated row sum at the end.
// ---------------------------------------------------------------------------
__global__ __launch_bounds__(128) void attn_fused(
    const __bf16* __restrict__ qd, const __bf16* __restrict__ kd,
    const __bf16* __restrict__ vt, const float* __restrict__ pi_q,
    const float* __restrict__ pi_k, __bf16* __restrict__ aout) {
    __shared__ __align__(16) __bf16 pbuf[4][16][VPAD];

    const int lane  = threadIdx.x & 31;
    const int wave  = threadIdx.x >> 5;
    const int bh    = blockIdx.x;                 // b*H + h
    const int b     = bh / Hhead, h = bh % Hhead;
    const int qbase = blockIdx.y * 64 + wave * 16;
    if (qbase >= Nseq) return;                    // whole wave out of range

    const __bf16* qp = qd + (size_t)bh * NPAD * Dhead;
    const __bf16* kp = kd + (size_t)bh * NPAD * Dhead;
    const __bf16* vp = vt + (size_t)bh * Dhead * VPAD;

    // zero the LDS tail columns (>= 208) of this wave's P buffer
#pragma unroll
    for (int z = 0; z < 8; ++z) {
        int idx = z * 32 + lane;                  // 256 tail elements
        pbuf[wave][idx >> 4][NPAD + (idx & 15)] = (__bf16)0.0f;
    }

    // Q tile as two A fragments (K = 0..31, 32..63); padded rows are zero.
    v16bf aq0 = load_a_frag(qp, Dhead, qbase, 0);
    v16bf aq1 = load_a_frag(qp, Dhead, qbase, 32);

    // |q|^2 per row (lane l and l^16 hold disjoint K halves of row l&15)
    float q2p = 0.f;
#pragma unroll
    for (int i = 0; i < 16; ++i) {
        float x = (float)aq0[i]; q2p += x * x;
        float y = (float)aq1[i]; q2p += y * y;
    }
    q2p += __shfl_xor(q2p, 16, 32);

    const int myrow = qbase + (lane & 15);
    const int rcl   = myrow < Nseq ? myrow : Nseq - 1;
    float piq = fabsf(pi_q[h * Nseq + rcl]);
    piq = piq > 1.f ? 1.f : piq;

    const int rOff = 8 * (lane >> 4);
    float rs[8] = {0, 0, 0, 0, 0, 0, 0, 0};

    for (int jt = 0; jt < 13; ++jt) {
        const int jb = jt * 16;
        v16bf bk0 = load_b_frag(kp, Dhead, jb, 0);
        v16bf bk1 = load_b_frag(kp, Dhead, jb, 32);
        v8f s = {};
        s = wmma_bf16(aq0, bk0, s);
        s = wmma_bf16(aq1, bk1, s);

        float k2p = 0.f;
#pragma unroll
        for (int i = 0; i < 16; ++i) {
            float x = (float)bk0[i]; k2p += x * x;
            float y = (float)bk1[i]; k2p += y * y;
        }
        k2p += __shfl_xor(k2p, 16, 32);           // lane holds |k_j|^2, j=jb+(l&15)

        const int j   = jb + (lane & 15);
        const int jcl = j < Nseq ? j : Nseq - 1;
        float pik = fabsf(pi_k[h * Nseq + jcl]);
        pik = pik > 1.f ? 1.f : pik;
        const float jok = (j < Nseq) ? 1.f : 0.f;

#pragma unroll
        for (int r = 0; r < 8; ++r) {
            const int src = r + rOff;             // row index of this C slot
            float q2v  = __shfl(q2p, src, 32);
            float piqv = __shfl(piq, src, 32);
            float d2 = q2v + k2p - 2.f * s[r];
            d2 = d2 < 0.f ? 0.f : d2;
            float p = __expf(-0.0625f * d2) * piqv * pik * jok;  // SCALE/2 = 1/16
            rs[r] += p;
            pbuf[wave][src][jb + (lane & 15)] = (__bf16)p;
        }
    }

    // row sums: xor-reduce across the 16 lanes of each half -> every lane
    // holds the sum for row (r + 8*(lane>>4)), exactly the C-layout slot.
#pragma unroll
    for (int r = 0; r < 8; ++r)
#pragma unroll
        for (int off = 1; off < 16; off <<= 1)
            rs[r] += __shfl_xor(rs[r], off, 32);

    // O = P * V : 7 K-fragments over the 224-wide P row, 4 d-tiles of V^T
    v8f o0 = {}, o1 = {}, o2 = {}, o3 = {};
    const __bf16* pw = (const __bf16*)&pbuf[wave][0][0];
    for (int kt = 0; kt < 7; ++kt) {
        v16bf ap = load_a_frag(pw, VPAD, 0, kt * 32);
        o0 = wmma_bf16(ap, load_b_frag(vp, VPAD, 0,  kt * 32), o0);
        o1 = wmma_bf16(ap, load_b_frag(vp, VPAD, 16, kt * 32), o1);
        o2 = wmma_bf16(ap, load_b_frag(vp, VPAD, 32, kt * 32), o2);
        o3 = wmma_bf16(ap, load_b_frag(vp, VPAD, 48, kt * 32), o3);
    }

    v8f oacc[4] = {o0, o1, o2, o3};
#pragma unroll
    for (int dt = 0; dt < 4; ++dt)
#pragma unroll
        for (int r = 0; r < 8; ++r) {
            int qrow = qbase + r + rOff;
            if (qrow < Nseq) {
                float val = oacc[dt][r] / (rs[r] + 1e-6f);
                aout[((size_t)(b * Nseq + qrow)) * Cdim + h * Dhead +
                     dt * 16 + (lane & 15)] = (__bf16)val;
            }
        }
}

// ---------------------------------------------------------------------------
// Kernel 3: output projection + bias, f32 out.  Same tiling/TDM as gemm_qkv.
// ---------------------------------------------------------------------------
__global__ __launch_bounds__(256) void gemm_proj(
    const __bf16* __restrict__ ain,  // [25216][768]
    const __bf16* __restrict__ wpb,  // [768][768] (N x K major)
    const float* __restrict__ bias, float* __restrict__ out) {
    __shared__ __align__(16) __bf16 wtile[64 * KHALF];        // 48 KB

    const int lane    = threadIdx.x & 31;
    const int wave    = threadIdx.x >> 5;
    const int rowBase = blockIdx.x * 128 + (wave & 3) * 32;
    const int colBase = blockIdx.y * 64 + (wave >> 2) * 32;
    const int colLoc  = (wave >> 2) * 32;

    v8f c00 = {}, c01 = {}, c10 = {}, c11 = {};
    for (int half = 0; half < 2; ++half) {
        if (threadIdx.x == 0)
            tdm_load_2d(wpb + (size_t)(blockIdx.y * 64) * Cdim + half * KHALF,
                        (uint32_t)(uintptr_t)(void*)wtile,
                        KHALF, 64, Cdim, 64, Cdim);
        if (threadIdx.x < 32) __builtin_amdgcn_s_wait_tensorcnt(0);
        __syncthreads();
        for (int kb = 0; kb < KHALF; kb += 32) {
            v16bf a0 = load_a_frag(ain, Cdim, rowBase, half * KHALF + kb);
            v16bf a1 = load_a_frag(ain, Cdim, rowBase + 16, half * KHALF + kb);
            v16bf b0 = load_b_frag(wtile, KHALF, colLoc, kb);
            v16bf b1 = load_b_frag(wtile, KHALF, colLoc + 16, kb);
            c00 = wmma_bf16(a0, b0, c00);
            c01 = wmma_bf16(a0, b1, c01);
            c10 = wmma_bf16(a1, b0, c10);
            c11 = wmma_bf16(a1, b1, c11);
        }
        __syncthreads();
    }

    const int rOff = 8 * (lane >> 4);
    const int nCol = lane & 15;
    v8f acc[4] = {c00, c01, c10, c11};
#pragma unroll
    for (int t = 0; t < 4; ++t) {
        const int rt = t >> 1, ct = t & 1;
#pragma unroll
        for (int r = 0; r < 8; ++r) {
            int R  = rowBase + rt * 16 + r + rOff;
            int Cc = colBase + ct * 16 + nCol;
            out[(size_t)R * Cdim + Cc] = acc[t][r] + bias[Cc];
        }
    }
}

// ---------------------------------------------------------------------------
// Launch
// ---------------------------------------------------------------------------
extern "C" void kernel_launch(void* const* d_in, const int* in_sizes, int n_in,
                              void* d_out, int out_size, void* d_ws,
                              size_t ws_size, hipStream_t stream) {
    const float* x      = (const float*)d_in[0];
    const float* qkv_w  = (const float*)d_in[1];
    const float* proj_w = (const float*)d_in[2];
    const float* proj_b = (const float*)d_in[3];
    const float* pi_q   = (const float*)d_in[4];
    const float* pi_k   = (const float*)d_in[5];
    float* out = (float*)d_out;

    char* ws = (char*)d_ws;
    size_t off = 0;
    auto carve = [&](size_t bytes) -> char* {
        char* p = ws + off;
        off += (bytes + 255) & ~(size_t)255;
        return p;
    };

    const size_t xbN    = (size_t)Mrows * Cdim;           // 19.4M
    const size_t wqkvN  = (size_t)3 * Cdim * Cdim;        // 1.77M
    const size_t wprojN = (size_t)Cdim * Cdim;            // 0.59M
    const size_t qN     = (size_t)Bsz * Hhead * NPAD * Dhead;
    const size_t vN     = (size_t)Bsz * Hhead * Dhead * VPAD;

    __bf16* xb    = (__bf16*)carve(xbN * 2);
    __bf16* wqkvb = (__bf16*)carve(wqkvN * 2);
    __bf16* wprojb= (__bf16*)carve(wprojN * 2);
    __bf16* qd    = (__bf16*)carve(qN * 2);
    __bf16* kd    = (__bf16*)carve(qN * 2);
    __bf16* vt    = (__bf16*)carve(vN * 2);
    __bf16* aout  = (__bf16*)carve(xbN * 2);

    // 1) converts
    cvt_f32_to_bf16<<<(int)((xbN + 255) / 256), 256, 0, stream>>>(x, xb, (int)xbN);
    cvt_f32_to_bf16<<<(int)((wqkvN + 255) / 256), 256, 0, stream>>>(qkv_w, wqkvb, (int)wqkvN);
    cvt_f32_to_bf16<<<(int)((wprojN + 255) / 256), 256, 0, stream>>>(proj_w, wprojb, (int)wprojN);

    // 2) zero padded q/k/vt (contiguous carve region, sizes are 256B multiples)
    long zeroN = (long)((qN * 2 + qN * 2 + vN * 2) / 16);
    zero_b128<<<(int)((zeroN + 255) / 256), 256, 0, stream>>>((uint4*)qd, zeroN);

    // 3) QKV projection GEMM (bf16 WMMA, TDM-staged weights) + scatter
    gemm_qkv<<<dim3(Mrows / 128, (3 * Cdim) / 64), 256, 0, stream>>>(
        xb, wqkvb, qd, kd, vt);

    // 4) fused GMM attention
    attn_fused<<<dim3(Bsz * Hhead, 4), 128, 0, stream>>>(
        qd, kd, vt, pi_q, pi_k, aout);

    // 5) output projection + bias (bf16 WMMA, TDM-staged weights)
    gemm_proj<<<dim3(Mrows / 128, Cdim / 64), 256, 0, stream>>>(
        aout, wprojb, proj_b, out);
}
